// UniPhyBlock_7370163880312
// MI455X (gfx1250) — matrix operand order repaired and checked by reading, not compiled, not run on hardware
//
#include <hip/hip_runtime.h>
#include <math.h>

// ---------------- problem dims ----------------
#define Bn 2
#define Tn 8
#define Cn 32
#define Hn 64
#define Wn 64
#define BTn (Bn*Tn)            // 16
#define HWn (Hn*Wn)            // 4096
#define CHWn (Cn*Hn*Wn)        // 131072
#define NCOMPLEX ((size_t)BTn*CHWn)   // 2097152 per re/im buffer

typedef __attribute__((ext_vector_type(16))) __bf16 v16bf;
typedef __attribute__((ext_vector_type(8)))  float  v8f;
typedef __attribute__((ext_vector_type(2)))  float  v2f;

// ================= Tensor Data Mover (gfx1250) =================
#if defined(__AMDGCN__) && __has_builtin(__builtin_amdgcn_tensor_load_to_lds) && \
    __has_builtin(__builtin_amdgcn_s_wait_tensorcnt)
#define HAVE_TDM 1
typedef __attribute__((ext_vector_type(4))) unsigned tdm_v4u;
typedef __attribute__((ext_vector_type(8))) int      tdm_v8i;
typedef __attribute__((ext_vector_type(4))) int      tdm_v4i;

__device__ __forceinline__ unsigned lds_addr_of(const void* p) {
  // generic (shared) pointer -> LDS byte offset
  return (unsigned)(unsigned long long)
         (__attribute__((address_space(3))) const void*)p;
}

// 2D f32 tile DMA: tile_d0 contiguous elements per row, tile_d1 rows,
// row stride tensor_stride0 (elements). tensor dims == tile dims (no OOB).
__device__ __forceinline__ void tdm_load_tile_f32(const float* gsrc,
                                                  unsigned lds_byte_addr,
                                                  unsigned tile_d0,
                                                  unsigned tile_d1,
                                                  unsigned tensor_stride0) {
  unsigned long long ga = (unsigned long long)gsrc;
  tdm_v4u g0;
  g0[0] = 1u;                                              // count=1 (valid)
  g0[1] = lds_byte_addr;                                   // lds_addr
  g0[2] = (unsigned)(ga & 0xffffffffu);                    // global_addr lo
  g0[3] = (unsigned)((ga >> 32) & 0x01ffffffu) | 0x80000000u; // hi | type=2
  tdm_v8i g1;
  g1[0] = 0x00020000;                    // data_size = 2 (4 bytes)
  g1[1] = (int)(tile_d0 << 16);          // tensor_dim0[15:0] in bits 63:48
  g1[2] = (int)(tile_d1 << 16);          // tensor_dim0 hi=0 | tensor_dim1 lo
  g1[3] = (int)(tile_d0 << 16);          // tensor_dim1 hi=0 | tile_dim0
  g1[4] = (int)tile_d1;                  // tile_dim1 | tile_dim2=0
  g1[5] = (int)tensor_stride0;           // tensor_dim0_stride lo32
  g1[6] = 0;
  g1[7] = 0;
  tdm_v4i gz4 = {0, 0, 0, 0};
#if __clang_major__ >= 23
  tdm_v8i gz8 = {0, 0, 0, 0, 0, 0, 0, 0};
  __builtin_amdgcn_tensor_load_to_lds(g0, g1, gz4, gz4, gz8, 0);
#else
  __builtin_amdgcn_tensor_load_to_lds(g0, g1, gz4, gz4, 0);
#endif
}
#endif  // HAVE_TDM

// ============ WMMA fragment helpers (wave32, CDNA5 layouts) ============
// A-matrix 16x32 bf16 (MxK): lanes0-15 -> M=lane, K in {0..7,16..23};
// lanes16-31 -> M=lane-16, K in {8..15,24..31}.
__device__ __forceinline__ v16bf load_a_bf(const float* A, int lda, int row0,
                                           int k0, float scale = 1.f) {
  int lane = threadIdx.x & 31;
  int m = row0 + (lane & 15);
  int kg = lane >> 4;
  v16bf out;
#pragma unroll
  for (int v = 0; v < 8; ++v) {
    int kl = (v < 4) ? (8 * kg + 2 * v) : (16 + 8 * kg + 2 * (v - 4));
    out[2 * v]     = (__bf16)(scale * A[m * lda + k0 + kl]);
    out[2 * v + 1] = (__bf16)(scale * A[m * lda + k0 + kl + 1]);
  }
  return out;
}
// A fragment when matrix is stored TRANSPOSED: A[m][k] at At[k*ldt + m]
__device__ __forceinline__ v16bf load_at_bf(const float* At, int ldt, int row0,
                                            int k0) {
  int lane = threadIdx.x & 31;
  int m = row0 + (lane & 15);
  int kg = lane >> 4;
  v16bf out;
#pragma unroll
  for (int v = 0; v < 8; ++v) {
    int kl = (v < 4) ? (8 * kg + 2 * v) : (16 + 8 * kg + 2 * (v - 4));
    out[2 * v]     = (__bf16)At[(k0 + kl) * ldt + m];
    out[2 * v + 1] = (__bf16)At[(k0 + kl + 1) * ldt + m];
  }
  return out;
}
__device__ __forceinline__ v16bf load_a_bf16src(const __bf16* A, int lda,
                                                int row0, int k0) {
  int lane = threadIdx.x & 31;
  int m = row0 + (lane & 15);
  int kg = lane >> 4;
  v16bf out;
#pragma unroll
  for (int v = 0; v < 8; ++v) {
    int kl = (v < 4) ? (8 * kg + 2 * v) : (16 + 8 * kg + 2 * (v - 4));
    out[2 * v]     = A[m * lda + k0 + kl];
    out[2 * v + 1] = A[m * lda + k0 + kl + 1];
  }
  return out;
}
// B-matrix 32x16 bf16 (KxN): lanes0-15 -> N=lane, K in {0..7,16..23}; lanes16-31 -> K in {8..15,24..31}.
__device__ __forceinline__ v16bf load_b_bf(const float* Bm, int ldb, int k0,
                                           int col0) {
  int lane = threadIdx.x & 31;
  int n = col0 + (lane & 15);
  int kg = lane >> 4;
  v16bf out;
#pragma unroll
  for (int v = 0; v < 8; ++v) {
    int kl = (v < 4) ? (8 * kg + 2 * v) : (16 + 8 * kg + 2 * (v - 4));
    out[2 * v]     = (__bf16)Bm[(k0 + kl) * ldb + n];
    out[2 * v + 1] = (__bf16)Bm[(k0 + kl + 1) * ldb + n];
  }
  return out;
}
// C/D 16x16 f32: lane n=lane&15, VGPR r -> M = r + 8*(lane>=16)
__device__ __forceinline__ void store_tile_f32(const v8f& c, float* C, int ldc,
                                               int row0, int col0) {
  int lane = threadIdx.x & 31;
  int n = col0 + (lane & 15);
  int m0 = row0 + (lane >> 4) * 8;
#pragma unroll
  for (int r = 0; r < 8; ++r) C[(m0 + r) * ldc + n] = c[r];
}

// ============ complex 64x64 tile matmul (DFT building block) ============
__device__ __forceinline__ void cmul64(const float* Ar, const float* Ai, int lda,
                                       const float* Br, const float* Bi, int ldb,
                                       int trow, int tcol, v8f& cr, v8f& ci) {
#if defined(__AMDGCN__) && __has_builtin(__builtin_amdgcn_wmma_f32_16x16x4_f32)
  int lane = threadIdx.x & 31;
  int m = trow * 16 + (lane & 15);
  int n = tcol * 16 + (lane & 15);
  int kb = (lane >> 4) * 2;   // A 16x4: lanes0-15 K={0,1}, lanes16-31 K={2,3}
#pragma unroll 4
  for (int k0 = 0; k0 < 64; k0 += 4) {
    v2f ar, ai, nai, br, bi;
    ar[0] = Ar[m * lda + k0 + kb];     ar[1] = Ar[m * lda + k0 + kb + 1];
    float a0 = Ai[m * lda + k0 + kb], a1 = Ai[m * lda + k0 + kb + 1];
    ai[0] = a0;  ai[1] = a1;  nai[0] = -a0; nai[1] = -a1;
    br[0] = Br[(k0 + kb) * ldb + n];   br[1] = Br[(k0 + kb + 1) * ldb + n];
    bi[0] = Bi[(k0 + kb) * ldb + n];   bi[1] = Bi[(k0 + kb + 1) * ldb + n];
    cr = __builtin_amdgcn_wmma_f32_16x16x4_f32(false, ar,  false, br, (short)0, cr, false, false);
    cr = __builtin_amdgcn_wmma_f32_16x16x4_f32(false, nai, false, bi, (short)0, cr, false, false);
    ci = __builtin_amdgcn_wmma_f32_16x16x4_f32(false, ar,  false, bi, (short)0, ci, false, false);
    ci = __builtin_amdgcn_wmma_f32_16x16x4_f32(false, ai,  false, br, (short)0, ci, false, false);
  }
#else
#pragma unroll
  for (int k0 = 0; k0 < 64; k0 += 32) {
    v16bf ar  = load_a_bf(Ar, lda, trow * 16, k0, 1.f);
    v16bf aip = load_a_bf(Ai, lda, trow * 16, k0, 1.f);
    v16bf ain = load_a_bf(Ai, lda, trow * 16, k0, -1.f);
    v16bf br  = load_b_bf(Br, ldb, k0, tcol * 16);
    v16bf bi  = load_b_bf(Bi, ldb, k0, tcol * 16);
    cr = __builtin_amdgcn_wmma_f32_16x16x32_bf16(false, ar,  false, br, (short)0, cr, false, false);
    cr = __builtin_amdgcn_wmma_f32_16x16x32_bf16(false, ain, false, bi, (short)0, cr, false, false);
    ci = __builtin_amdgcn_wmma_f32_16x16x32_bf16(false, ar,  false, bi, (short)0, ci, false, false);
    ci = __builtin_amdgcn_wmma_f32_16x16x32_bf16(false, aip, false, br, (short)0, ci, false, false);
  }
#endif
}

// ============ small math helpers ============
__device__ __forceinline__ float gelu_tanh(float x) {
  float x3 = x * x * x;
  return 0.5f * x * (1.f + tanhf(0.79788456080286535588f * (x + 0.044715f * x3)));
}
__device__ __forceinline__ unsigned pcg(unsigned v) {
  v = v * 747796405u + 2891336453u;
  unsigned w = ((v >> ((v >> 28u) + 4u)) ^ v) * 277803737u;
  return (w >> 22u) ^ w;
}
__device__ __forceinline__ void gauss2(unsigned s, float* g0, float* g1) {
  unsigned a = pcg(s * 2u + 0x9E3779B9u), b = pcg(s * 2u + 0x85EBCA6Bu);
  float u1 = ((float)a + 1.f) * 2.3283064365386963e-10f;  // (0,1]
  float u2 = (float)b * 2.3283064365386963e-10f;
  float r = sqrtf(-2.f * logf(u1));
  *g0 = r * cosf(6.283185307179586f * u2);
  *g1 = r * sinf(6.283185307179586f * u2);
}

// ============ kernel 0: DFT twiddle tables ============
__global__ void dft_init_kernel(float* Fr, float* Fi, float* Gr, float* Gi) {
  int i = blockIdx.x * blockDim.x + threadIdx.x;
  if (i >= 64 * 64) return;
  int r = i >> 6, c = i & 63;
  float ang = -6.283185307179586f * (float)((r * c) & 63) * (1.f / 64.f);
  float cs = cosf(ang), sn = sinf(ang);
  Fr[i] = cs;                 // forward: exp(-i 2pi rc/64)
  Fi[i] = sn;
  Gr[i] = cs * (1.f / 64.f);  // inverse: exp(+i 2pi rc/64)/64 per dim
  Gi[i] = -sn * (1.f / 64.f);
}

// ============ kernel 1/5: channel layernorm over 2C at each position ======
__global__ void ln_kernel(const float* __restrict__ xr, const float* __restrict__ xi,
                          const float* __restrict__ g, const float* __restrict__ b,
                          float* __restrict__ yr, float* __restrict__ yi) {
  int pos = blockIdx.x * blockDim.x + threadIdx.x;  // (bt,h,w)
  if (pos >= BTn * HWn) return;
  int bt = pos / HWn, hw = pos - bt * HWn;
  const float* pr = xr + (size_t)bt * CHWn + hw;
  const float* pi = xi + (size_t)bt * CHWn + hw;
  float s = 0.f, s2 = 0.f;
#pragma unroll 8
  for (int c = 0; c < Cn; ++c) {
    float a = pr[c * HWn], d = pi[c * HWn];
    s += a + d;
    s2 += a * a + d * d;
  }
  float mu = s * (1.f / (2 * Cn));
  float var = s2 * (1.f / (2 * Cn)) - mu * mu;
  float rinv = rsqrtf(var + 1e-5f);
#pragma unroll 8
  for (int c = 0; c < Cn; ++c) {
    float a = pr[c * HWn], d = pi[c * HWn];
    yr[(size_t)bt * CHWn + c * HWn + hw] = (a - mu) * rinv * g[c] + b[c];
    yi[(size_t)bt * CHWn + c * HWn + hw] = (d - mu) * rinv * g[Cn + c] + b[Cn + c];
  }
}

// ============ kernel 2: Clifford conv via implicit-GEMM bf16 WMMA ========
// K ordered as k = kh*192 + kw*64 + ic   (K = 576 = 18 * 32)
__global__ void conv_wmma_kernel(const float* __restrict__ xr,
                                 const float* __restrict__ xi,
                                 const float* __restrict__ w,
                                 const float* __restrict__ bias,
                                 float* __restrict__ outr,
                                 float* __restrict__ outi) {
  int pixt = blockIdx.x;     // 0..255 : 16 pixels each
  int oct  = blockIdx.y;     // 0..3   : 16 out-channels each
  int img  = blockIdx.z;     // 0..15
  int lane = threadIdx.x & 31;
  int pix0 = pixt * 16;
  v8f acc = {};
  int mA  = oct * 16 + (lane & 15);
  int kgA = lane >> 4;
  int ncol = pix0 + (lane & 15);
  int hB = ncol >> 6, wB = ncol & 63;
  int khalfB = (lane >> 4) * 16;
  for (int kb = 0; kb < 18; ++kb) {
    v16bf a, bf;
#pragma unroll
    for (int v = 0; v < 8; ++v) {
      int kl = (v < 4) ? (8 * kgA + 2 * v) : (16 + 8 * kgA + 2 * (v - 4));
#pragma unroll
      for (int e = 0; e < 2; ++e) {
        int k = kb * 32 + kl + e;
        int kh = k / 192, rem = k - kh * 192;
        int kw = rem >> 6, ic = rem & 63;
        a[2 * v + e] = (__bf16)w[((mA * 64 + ic) * 3 + kh) * 3 + kw];
      }
    }
#pragma unroll
    for (int v = 0; v < 8; ++v) {
#pragma unroll
      for (int e = 0; e < 2; ++e) {
        int k = kb * 32 + ((v < 4) ? (khalfB / 2 + 2 * v) : (16 + khalfB / 2 + 2 * (v - 4))) + e;
        int kh = k / 192, rem = k - kh * 192;
        int kw = rem >> 6, ic = rem & 63;
        int h2 = hB + kh - 1, w2 = wB + kw - 1;
        float val = 0.f;
        if (h2 >= 0 && h2 < 64 && w2 >= 0 && w2 < 64) {
          const float* src = (ic < Cn) ? xr : xi;
          val = src[(size_t)img * CHWn + (ic & (Cn - 1)) * HWn + h2 * 64 + w2];
        }
        bf[2 * v + e] = (__bf16)val;
      }
    }
    acc = __builtin_amdgcn_wmma_f32_16x16x32_bf16(false, a, false, bf, (short)0,
                                                  acc, false, false);
  }
  int nn = lane & 15;
  int m0 = oct * 16 + (lane >> 4) * 8;
#pragma unroll
  for (int r = 0; r < 8; ++r) {
    int oc = m0 + r;
    float val = acc[r] + bias[oc];
    float* dst = (oc < Cn) ? outr : outi;
    dst[(size_t)img * CHWn + (oc & (Cn - 1)) * HWn + pix0 + nn] = val;
  }
}

// ============ kernel 3: spectral branch — fft2 as 4 complex DFT GEMMs ====
// one 256-thread workgroup per (bt,c) image; image staged to LDS by the TDM.
__global__ void spectral_kernel(const float* __restrict__ xr,
                                const float* __restrict__ xi,
                                const float* __restrict__ Fr, const float* __restrict__ Fi,
                                const float* __restrict__ Gr, const float* __restrict__ Gi,
                                const float* __restrict__ wr, const float* __restrict__ wi,
                                float* __restrict__ outr, float* __restrict__ outi) {
  __shared__ float Xr[64 * 64], Xi_[64 * 64], Ur[64 * 64], Ui[64 * 64];  // 64 KB
  int img = blockIdx.x;            // bt*C + c
  int ch = img & (Cn - 1);
  int tid = threadIdx.x;
#if HAVE_TDM
  if (tid < 32) {  // wave 0 issues both DMAs (TDM ignores EXEC, 1 issue/wave)
    tdm_load_tile_f32(xr + (size_t)img * HWn, lds_addr_of(Xr), 64, 64, 64);
    tdm_load_tile_f32(xi + (size_t)img * HWn, lds_addr_of(Xi_), 64, 64, 64);
    __builtin_amdgcn_s_wait_tensorcnt(0);
  }
#else
  for (int i = tid; i < 64 * 64; i += 256) {
    Xr[i]  = xr[(size_t)img * HWn + i];
    Xi_[i] = xi[(size_t)img * HWn + i];
  }
#endif
  __syncthreads();
  int wave = tid >> 5;
  int lane = tid & 31;
  // pass 1: U = F @ X   (transforms H axis)
  for (int t = wave; t < 16; t += 8) {
    int tr = t >> 2, tc = t & 3;
    v8f cr = {}, ci = {};
    cmul64(Fr, Fi, 64, Xr, Xi_, 64, tr, tc, cr, ci);
    store_tile_f32(cr, Ur, 64, tr * 16, tc * 16);
    store_tile_f32(ci, Ui, 64, tr * 16, tc * 16);
  }
  __syncthreads();
  // pass 2: X = (U @ F) * specW   (transforms W axis, apply filter)
  for (int t = wave; t < 16; t += 8) {
    int tr = t >> 2, tc = t & 3;
    v8f cr = {}, ci = {};
    cmul64(Ur, Ui, 64, Fr, Fi, 64, tr, tc, cr, ci);
    int n = tc * 16 + (lane & 15);
    int m0 = tr * 16 + (lane >> 4) * 8;
#pragma unroll
    for (int r = 0; r < 8; ++r) {
      int m = m0 + r;
      float fr = wr[(size_t)ch * HWn + m * 64 + n];
      float fi = wi[(size_t)ch * HWn + m * 64 + n];
      Xr[m * 64 + n]  = cr[r] * fr - ci[r] * fi;
      Xi_[m * 64 + n] = cr[r] * fi + ci[r] * fr;
    }
  }
  __syncthreads();
  // pass 3: U = G @ X  (inverse over H axis, includes 1/64)
  for (int t = wave; t < 16; t += 8) {
    int tr = t >> 2, tc = t & 3;
    v8f cr = {}, ci = {};
    cmul64(Gr, Gi, 64, Xr, Xi_, 64, tr, tc, cr, ci);
    store_tile_f32(cr, Ur, 64, tr * 16, tc * 16);
    store_tile_f32(ci, Ui, 64, tr * 16, tc * 16);
  }
  __syncthreads();
  // pass 4: out = U @ G  (inverse over W axis) -> global
  for (int t = wave; t < 16; t += 8) {
    int tr = t >> 2, tc = t & 3;
    v8f cr = {}, ci = {};
    cmul64(Ur, Ui, 64, Gr, Gi, 64, tr, tc, cr, ci);
    int n = tc * 16 + (lane & 15);
    int m0 = tr * 16 + (lane >> 4) * 8;
#pragma unroll
    for (int r = 0; r < 8; ++r) {
      outr[(size_t)img * HWn + (m0 + r) * 64 + n] = cr[r];
      outi[(size_t)img * HWn + (m0 + r) * 64 + n] = ci[r];
    }
  }
}

// ============ kernel 4: gate mix + residual ============
__global__ void gate_kernel(const float* __restrict__ cr, const float* __restrict__ ci,
                            const float* __restrict__ sr, const float* __restrict__ si,
                            const float* __restrict__ xr, const float* __restrict__ xi,
                            const float* __restrict__ gate,
                            float* __restrict__ zr, float* __restrict__ zi) {
  size_t i = (size_t)blockIdx.x * blockDim.x + threadIdx.x;
  if (i >= NCOMPLEX) return;
  float g = gate[0];
  zr[i] = g * cr[i] + (1.f - g) * sr[i] + xr[i];
  zi[i] = g * ci[i] + (1.f - g) * si[i] + xi[i];
}

// ============ kernel 6: ctx = mean |znc| over (H,W) ============
__global__ void ctx_kernel(const float* __restrict__ zr, const float* __restrict__ zi,
                           float* __restrict__ ctx) {
  __shared__ float red[256];
  int btc = blockIdx.x;  // (bt,c)
  float s = 0.f;
  for (int i = threadIdx.x; i < HWn; i += 256) {
    float a = zr[(size_t)btc * HWn + i], b = zi[(size_t)btc * HWn + i];
    s += sqrtf(a * a + b * b);
  }
  red[threadIdx.x] = s;
  __syncthreads();
  for (int off = 128; off > 0; off >>= 1) {
    if (threadIdx.x < off) red[threadIdx.x] += red[threadIdx.x + off];
    __syncthreads();
  }
  if (threadIdx.x == 0) ctx[btc] = red[0] * (1.f / (float)HWn);
}

// ============ kernel 7: input gain + evolution operator ============
__global__ void gainevo_kernel(const float* __restrict__ ctx,
                               const float* __restrict__ gW, const float* __restrict__ gb,
                               const float* __restrict__ alpha, const float* __restrict__ omega,
                               const float* __restrict__ dt,
                               float* __restrict__ gain, float* __restrict__ evr,
                               float* __restrict__ evi) {
  int idx = blockIdx.x * blockDim.x + threadIdx.x;
  if (idx >= BTn * Cn) return;
  int bt = idx / Cn, c = idx - bt * Cn;
  float s = gb[c];
#pragma unroll 8
  for (int k = 0; k < Cn; ++k) s += ctx[bt * Cn + k] * gW[k * Cn + c];
  gain[idx] = 1.f / (1.f + expf(-s));
  float a = alpha[c];
  float sp = (a > 20.f) ? a : log1pf(expf(a));   // softplus
  float d = dt[bt];
  float mag = expf(-d * sp);
  float ph = d * omega[c];
  evr[idx] = mag * cosf(ph);
  evi[idx] = mag * sinf(ph);
}

// ============ kernel 8: temporal scan (T=8) + noise + residual ============
__global__ void scan_kernel(const float* __restrict__ znr, const float* __restrict__ zni,
                            const float* __restrict__ zr,  const float* __restrict__ zi,
                            const float* __restrict__ gain,
                            const float* __restrict__ evr, const float* __restrict__ evi,
                            const float* __restrict__ dt,  const float* __restrict__ sigma,
                            float* __restrict__ x2r, float* __restrict__ x2i) {
  int idx = blockIdx.x * blockDim.x + threadIdx.x;  // (b, c, hw)
  if (idx >= Bn * Cn * HWn) return;
  int b = idx / (Cn * HWn);
  int rem = idx - b * (Cn * HWn);
  int c = rem / HWn, hw = rem - c * HWn;
  float hr = 0.f, hi = 0.f;
#pragma unroll
  for (int t = 0; t < Tn; ++t) {
    int bt = b * Tn + t;
    size_t off = ((size_t)bt * Cn + c) * HWn + hw;
    float g = gain[bt * Cn + c];
    float ur = znr[off] * g, ui = zni[off] * g;
    float ar = evr[bt * Cn + c], ai = evi[bt * Cn + c];
    float nhr = ar * hr - ai * hi + ur;
    float nhi = ar * hi + ai * hr + ui;
    hr = nhr; hi = nhi;
    float sc = sigma[c] * sqrtf(dt[bt] + 1e-6f);
    float er, ei;
    gauss2((unsigned)off, &er, &ei);
    x2r[off] = hr + sc * er + zr[off];
    x2i[off] = hi + sc * ei + zi[off];
  }
}

// ============ kernel 9: channelwise MLP (64 -> 256 gelu -> 64) + residual ==
// one block = 64 rows; activation tile staged feature-major in LDS by the TDM
// (64 contiguous pixels x 32 channels, stride H*W), WMMA GEMMs on bf16.
__global__ void mlp_wmma_kernel(const float* __restrict__ x2r,
                                const float* __restrict__ x2i,
                                const float* __restrict__ pw1, const float* __restrict__ pb1,
                                const float* __restrict__ pw2, const float* __restrict__ pb2,
                                float* __restrict__ out) {
  __shared__ float  Xs[64 * 64];     // 16 KB, TRANSPOSED: Xs[f*64 + row]
  __shared__ __bf16 Hs[64 * 256];    // 32 KB bf16 hidden
  int tid = threadIdx.x;
  int row0 = blockIdx.x * 64;        // 64 rows share bt; hw contiguous
#if HAVE_TDM
  if (tid < 32) {
    int bt = row0 >> 12, hw0 = row0 & 4095;
    tdm_load_tile_f32(x2r + (size_t)bt * CHWn + hw0, lds_addr_of(Xs),
                      64, 32, HWn);
    tdm_load_tile_f32(x2i + (size_t)bt * CHWn + hw0, lds_addr_of(Xs + 32 * 64),
                      64, 32, HWn);
    __builtin_amdgcn_s_wait_tensorcnt(0);
  }
#else
  for (int i = tid; i < 64 * 64; i += 256) {
    int f = i >> 6, rl = i & 63;
    int row = row0 + rl;
    int bt = row >> 12, hw = row & 4095;
    const float* src = (f < Cn) ? x2r : x2i;
    Xs[f * 64 + rl] = src[((size_t)bt * Cn + (f & (Cn - 1))) * HWn + hw];
  }
#endif
  __syncthreads();
  int wave = tid >> 5, lane = tid & 31;
  // ---- GEMM1: (64x64) @ (64x256) , gelu -> Hs ----
  for (int t = wave; t < 64; t += 8) {
    int tr = t >> 4, tc = t & 15;
    v8f acc = {};
#pragma unroll
    for (int k0 = 0; k0 < 64; k0 += 32) {
      if (k0 + 32 < 64) __builtin_prefetch(pw1 + (k0 + 32) * 256 + tc * 16, 0, 3);
      v16bf a = load_at_bf(Xs, 64, tr * 16, k0);   // transposed LDS layout
      v16bf bfr = load_b_bf(pw1, 256, k0, tc * 16);
      acc = __builtin_amdgcn_wmma_f32_16x16x32_bf16(false, a, false, bfr,
                                                    (short)0, acc, false, false);
    }
    int n = tc * 16 + (lane & 15);
    int m0 = tr * 16 + (lane >> 4) * 8;
    float bias = pb1[n];
#pragma unroll
    for (int r = 0; r < 8; ++r)
      Hs[(m0 + r) * 256 + n] = (__bf16)gelu_tanh(acc[r] + bias);
  }
  __syncthreads();
  // ---- GEMM2: (64x256) @ (256x64) + bias + residual -> out ----
  for (int t = wave; t < 16; t += 8) {
    int tr = t >> 2, tc = t & 3;
    v8f acc = {};
#pragma unroll
    for (int k0 = 0; k0 < 256; k0 += 32) {
      __builtin_prefetch(pw2 + (k0 + 32) * 64, 0, 3);
      v16bf a = load_a_bf16src(Hs, 256, tr * 16, k0);
      v16bf bfr = load_b_bf(pw2, 64, k0, tc * 16);
      acc = __builtin_amdgcn_wmma_f32_16x16x32_bf16(false, a, false, bfr,
                                                    (short)0, acc, false, false);
    }
    int n = tc * 16 + (lane & 15);     // output feature 0..63
    int m0 = tr * 16 + (lane >> 4) * 8;
    float bias = pb2[n];
#pragma unroll
    for (int r = 0; r < 8; ++r) {
      int row = row0 + m0 + r;
      int bt = row >> 12, hw = row & 4095;
      const float* res = (n < Cn) ? x2r : x2i;
      float v = acc[r] + bias +
                res[((size_t)bt * Cn + (n & (Cn - 1))) * HWn + hw];
      // output (B,T,2C,H,W): channel n<32 real, n>=32 imag
      out[((size_t)bt * (2 * Cn) + n) * HWn + hw] = v;
    }
  }
}

// ================= host-side orchestration =================
extern "C" void kernel_launch(void* const* d_in, const int* in_sizes, int n_in,
                              void* d_out, int out_size, void* d_ws, size_t ws_size,
                              hipStream_t stream) {
  const float* x_r    = (const float*)d_in[0];
  const float* x_i    = (const float*)d_in[1];
  const float* dt     = (const float*)d_in[2];
  const float* ln_s_g = (const float*)d_in[3];
  const float* ln_s_b = (const float*)d_in[4];
  const float* cliffw = (const float*)d_in[5];
  const float* cliffb = (const float*)d_in[6];
  const float* specwr = (const float*)d_in[7];
  const float* specwi = (const float*)d_in[8];
  const float* gate   = (const float*)d_in[9];
  const float* ln_t_g = (const float*)d_in[10];
  const float* ln_t_b = (const float*)d_in[11];
  const float* alpha  = (const float*)d_in[12];
  const float* omega  = (const float*)d_in[13];
  const float* gain_W = (const float*)d_in[14];
  const float* gain_b = (const float*)d_in[15];
  const float* sigma  = (const float*)d_in[16];
  const float* pw1    = (const float*)d_in[17];
  const float* pb1    = (const float*)d_in[18];
  const float* pw2    = (const float*)d_in[19];
  const float* pb2    = (const float*)d_in[20];
  float* out = (float*)d_out;

  // workspace carve-up (floats)
  float* ws = (float*)d_ws;
  const size_t N = NCOMPLEX;
  float* Ar = ws;            float* Ai = Ar + N;   // xnc, later z
  float* Br = Ai + N;        float* Bi = Br + N;   // cliff, later znc
  float* Cr = Bi + N;        float* Ci = Cr + N;   // spec, later x2
  float* Fr = Ci + N;        float* Fi = Fr + 4096;
  float* Gr = Fi + 4096;     float* Gi = Gr + 4096;
  float* ctx  = Gi + 4096;
  float* gain = ctx + BTn * Cn;
  float* evr  = gain + BTn * Cn;
  float* evi  = evr + BTn * Cn;

  dft_init_kernel<<<16, 256, 0, stream>>>(Fr, Fi, Gr, Gi);
  ln_kernel<<<BTn * HWn / 256, 256, 0, stream>>>(x_r, x_i, ln_s_g, ln_s_b, Ar, Ai);
  conv_wmma_kernel<<<dim3(256, 4, BTn), 32, 0, stream>>>(Ar, Ai, cliffw, cliffb, Br, Bi);
  spectral_kernel<<<BTn * Cn, 256, 0, stream>>>(Ar, Ai, Fr, Fi, Gr, Gi,
                                                specwr, specwi, Cr, Ci);
  gate_kernel<<<(int)(N / 256), 256, 0, stream>>>(Br, Bi, Cr, Ci, x_r, x_i, gate, Ar, Ai);
  ln_kernel<<<BTn * HWn / 256, 256, 0, stream>>>(Ar, Ai, ln_t_g, ln_t_b, Br, Bi);
  ctx_kernel<<<BTn * Cn, 256, 0, stream>>>(Br, Bi, ctx);
  gainevo_kernel<<<2, 256, 0, stream>>>(ctx, gain_W, gain_b, alpha, omega, dt,
                                        gain, evr, evi);
  scan_kernel<<<(Bn * Cn * HWn) / 256, 256, 0, stream>>>(Br, Bi, Ar, Ai, gain,
                                                         evr, evi, dt, sigma, Cr, Ci);
  mlp_wmma_kernel<<<(BTn * HWn) / 64, 256, 0, stream>>>(Cr, Ci, pw1, pb1, pw2, pb2, out);
  (void)in_sizes; (void)n_in; (void)out_size; (void)ws_size;
}
